// MemoryGraph_78108275245234
// MI455X (gfx1250) — compile-verified, matrix operand order checked by reading, not run on hardware
//
#include <hip/hip_runtime.h>
#include <hip/hip_bf16.h>

#define BSZ   8
#define NCC   1024
#define CNN   64
#define DNN   64
#define HSS   256
#define HMM   256
#define HMODD 128
#define MODIN 198
#define MODOUT 68

typedef __bf16 bf16_t;
typedef bf16_t v16bf __attribute__((ext_vector_type(16)));
typedef float  v8f   __attribute__((ext_vector_type(8)));

__device__ __forceinline__ v8f wmma_bf16(v16bf a, v16bf b, v8f c) {
  return __builtin_amdgcn_wmma_f32_16x16x32_bf16(false, a, false, b, (short)0, c, false, false);
}

// A fragment from row-major [M][K] (bf16), tile origin p=&A[m0][k0], leading dim ld.
// lane<16: m=lane, K 0-7 & 16-23; lane>=16: m=lane-16, K 8-15 & 24-31.
__device__ __forceinline__ v16bf fragA_rm(const bf16_t* p, int ld, int lane) {
  const bf16_t* row = p + (lane & 15) * ld + (lane >> 4) * 8;
  v16bf f;
#pragma unroll
  for (int j = 0; j < 16; ++j) {
    int kl = (j < 8) ? j : (j + 8);
    f[j] = row[kl];
  }
  return f;
}

// Fragment-packed B layout: for K-chunk kc, khalf, column n, the 16 bf16 values a
// lane needs are contiguous at ((kc*2+khalf)*N + n)*16.  Element j holds
// k = kc*32 + khalf*8 + ((j<8)?j:j+8).
__device__ __forceinline__ int pack_off(int k, int n, int N) {
  int kc = k >> 5, kl = k & 31;
  int khalf = (kl >> 3) & 1;
  int j = (kl & 7) | ((kl & 16) >> 1);
  return ((kc * 2 + khalf) * N + n) * 16 + j;
}

// B fragment from packed layout (contiguous 32B per lane, coalesced across lanes).
__device__ __forceinline__ v16bf fragB_pk(const bf16_t* p, int N, int kc, int n0, int lane) {
  const bf16_t* q = p + ((size_t)(kc * 2 + (lane >> 4)) * N + (n0 + (lane & 15))) * 16;
  v16bf f;
#pragma unroll
  for (int j = 0; j < 16; ++j) f[j] = q[j];
  return f;
}

// B fragment from f32 [K][N] with bounds (zero-pad), converting to bf16.
__device__ __forceinline__ v16bf fragB_f32(const float* p, int ldn, int k0, int n0,
                                           int kmax, int nmax, int lane) {
  int n  = n0 + (lane & 15);
  int kb = k0 + (lane >> 4) * 8;
  v16bf f;
#pragma unroll
  for (int j = 0; j < 16; ++j) {
    int k = kb + ((j < 8) ? j : (j + 8));
    float v = (k < kmax && n < nmax) ? p[(size_t)k * ldn + n] : 0.0f;
    f[j] = (bf16_t)v;
  }
  return f;
}

__device__ __forceinline__ float sigm(float x) { return 1.0f / (1.0f + expf(-x)); }

// ---------------- prep: bf16 + fragment-pack shared MLP weights into ws ----------------
__global__ __launch_bounds__(256) void memgraph_prep_kernel(
    const float* sw1, const float* sw2, const float* mw1, const float* mw2,
    bf16_t* w1T, bf16_t* w2T, bf16_t* mw1T, bf16_t* mw2T) {
  int i = blockIdx.x * 256 + threadIdx.x;
  if (i < 32768) {                       // state_w1 [256 out][128 in] -> packed K=128,N=256
    int k = i >> 8, n = i & 255;
    w1T[pack_off(k, n, 256)] = (bf16_t)sw1[n * 128 + k];
  } else if (i < 49152) {                // state_w2 [64 out][256 in] -> packed K=256,N=64
    int j = i - 32768; int k = j >> 6, n = j & 63;
    w2T[pack_off(k, n, 64)] = (bf16_t)sw2[n * 256 + k];
  } else if (i < 65536) {                // msg_w1 [256 out][64 in] -> packed K=64,N=256
    int j = i - 49152; int k = j >> 8, n = j & 255;
    mw1T[pack_off(k, n, 256)] = (bf16_t)mw1[n * 64 + k];
  } else if (i < 81920) {                // msg_w2 [64 out][256 in] -> packed K=256,N=64
    int j = i - 65536; int k = j >> 6, n = j & 63;
    mw2T[pack_off(k, n, 64)] = (bf16_t)mw2[n * 256 + k];
  }
}

// ---------------- per-cell kernel: one block per (b,c) ----------------
__global__ __launch_bounds__(256) void memgraph_cell_kernel(
    const float* __restrict__ x, const float* __restrict__ h,
    const float* __restrict__ msg, const float* __restrict__ W,
    const float* __restrict__ decay_logit, const float* __restrict__ border_gate_logit,
    const float* __restrict__ neuron_id,
    const float* __restrict__ state_b1, const float* __restrict__ state_gs1, const float* __restrict__ state_gb1,
    const float* __restrict__ state_b2, const float* __restrict__ state_gs2, const float* __restrict__ state_gb2,
    const float* __restrict__ msg_b1, const float* __restrict__ msg_gs1, const float* __restrict__ msg_gb1,
    const float* __restrict__ msg_b2, const float* __restrict__ msg_gs2, const float* __restrict__ msg_gb2,
    const float* __restrict__ inject_w, const float* __restrict__ inject_b,
    const int* __restrict__ c2g,
    const bf16_t* __restrict__ w1T, const bf16_t* __restrict__ w2T,
    const bf16_t* __restrict__ mw1T, const bf16_t* __restrict__ mw2T,
    float* __restrict__ out_readout, float* __restrict__ out_h,
    float* __restrict__ out_msg, float* __restrict__ feats_ws) {
  __shared__ __attribute__((aligned(32))) unsigned char smem[65536];
  // region1 [0,16K): sW(8K)+sMsgP(8K, fragment-packed) -> later sAct bf16 [64][128]
  // region2 [16K,32K): sRecv f32 [64][64] -> later sHnew bf16[64][64] + sHm/sMm
  // region3 [32K,64K): sX f32[64] early -> later sHid bf16 [64][256]
  bf16_t* sW    = (bf16_t*)smem;
  bf16_t* sMsgP = (bf16_t*)(smem + 8192);
  bf16_t* sAct  = (bf16_t*)smem;
  float*  sRecv = (float*)(smem + 16384);
  bf16_t* sHnew = (bf16_t*)(smem + 16384);
  float*  sHm   = (float*)(smem + 16384 + 8192);
  float*  sMm   = (float*)(smem + 16384 + 8448);
  float*  sX    = (float*)(smem + 32768);
  bf16_t* sHid  = (bf16_t*)(smem + 32768);

  const int tid  = threadIdx.x;
  const int lane = tid & 31;
  const int wave = tid >> 5;
  const int b = blockIdx.x / NCC;
  const int c = blockIdx.x % NCC;
  const int g = c2g[c];
  const size_t cell = (size_t)b * NCC + c;

  const float* Wp = W   + cell * 4096;
  const float* Mp = msg + cell * 4096;
  const float* Hp = h   + cell * 4096;
  const float* Np = neuron_id + (size_t)c * 4096;

  // Prefetch h (phases 3 & 5) and neuron_id (phase 7): one 64B line per thread.
  {
    int i = tid * 16;
    __builtin_prefetch(Hp + i, 0, 3);
    __builtin_prefetch(Np + i, 0, 3);
  }

  // Phase 0: stage W as bf16 row-major; msg as bf16 fragment-packed [K=64][N=64]
  for (int i = tid; i < 4096; i += 256) {
    sW[i] = (bf16_t)Wp[i];
    int m = i >> 6, d = i & 63;
    sMsgP[pack_off(m, d, 64)] = (bf16_t)Mp[i];
  }
  if (tid < 64) sX[tid] = x[(size_t)b * (NCC * DNN) + c * 64 + tid];
  __syncthreads();

  // Phase 1: received = W @ msg  (M=N=K=64)
  for (int t = wave; t < 16; t += 8) {
    int mt = t >> 2, nt = t & 3;
    v8f acc = {};
#pragma unroll
    for (int kc = 0; kc < 2; ++kc) {
      v16bf a  = fragA_rm(sW + mt * 16 * 64 + kc * 32, 64, lane);
      v16bf bb = fragB_pk(sMsgP, 64, kc, nt * 16, lane);
      acc = wmma_bf16(a, bb, acc);
    }
    int mbase = mt * 16 + ((lane >> 4) << 3);
    int n = nt * 16 + (lane & 15);
#pragma unroll
    for (int r = 0; r < 8; ++r) sRecv[(mbase + r) * 64 + n] = acc[r];
  }
  __syncthreads();

  // Phase 2: inject into rows 0..3 and gated border exchange into rows 8..11
  {
    int o = tid;  // o = a*64 + d, rows 0..3
    const float* iw = inject_w + ((size_t)g * 256 + o) * 64;
    float s = inject_b[g * 256 + o];
#pragma unroll 8
    for (int i = 0; i < 64; ++i) s += sX[i] * iw[i];
    sRecv[o] += s;
  }
  {
    int p = tid >> 6, d = tid & 63;
    int gy = c >> 5, gx = c & 31;
    int src_c = -1, src_port = 0;
    if (p == 0 && gy > 0)       { src_c = c - 32; src_port = 1; }
    else if (p == 1 && gy < 31) { src_c = c + 32; src_port = 0; }
    else if (p == 2 && gx > 0)  { src_c = c - 1;  src_port = 3; }
    else if (p == 3 && gx < 31) { src_c = c + 1;  src_port = 2; }
    if (src_c >= 0) {
      float inc  = msg[(((size_t)b * NCC + src_c) * CNN + (8 + src_port)) * DNN + d];
      float gate = sigm(border_gate_logit[cell * 4 + p]);
      sRecv[(8 + p) * 64 + d] += gate * inc;
    }
  }
  __syncthreads();

  // Phase 3: build state-MLP input [64 tokens][128] = [received | h] as bf16
  for (int i = tid; i < 4096; i += 256) {
    int n = i >> 6, d = i & 63;
    sAct[n * 128 + d]      = (bf16_t)sRecv[i];
    sAct[n * 128 + 64 + d] = (bf16_t)Hp[i];
  }
  __syncthreads();

  // Phase 4: state layer1 (K=128, N=256) -> tanh affine -> sHid
  if (tid < 64) { sHm[tid] = 0.0f; sMm[tid] = 0.0f; }
  for (int t = wave; t < 64; t += 8) {
    int mt = t >> 4, nt = t & 15;
    v8f acc = {};
#pragma unroll
    for (int kc = 0; kc < 4; ++kc) {
      v16bf a  = fragA_rm(sAct + mt * 16 * 128 + kc * 32, 128, lane);
      v16bf bb = fragB_pk(w1T, 256, kc, nt * 16, lane);
      acc = wmma_bf16(a, bb, acc);
    }
    int n = nt * 16 + (lane & 15);
    float bv = state_b1[n], gsv = state_gs1[g * 256 + n], gbv = state_gb1[g * 256 + n];
    int mbase = mt * 16 + ((lane >> 4) << 3);
#pragma unroll
    for (int r = 0; r < 8; ++r)
      sHid[(mbase + r) * 256 + n] = (bf16_t)tanhf((acc[r] + bv) * gsv + gbv);
  }
  __syncthreads();

  // Phase 5: state layer2 (K=256, N=64) -> tanh affine -> decay blend -> h_new
  for (int t = wave; t < 16; t += 8) {
    int mt = t >> 2, nt = t & 3;
    v8f acc = {};
#pragma unroll
    for (int kc = 0; kc < 8; ++kc) {
      v16bf a  = fragA_rm(sHid + mt * 16 * 256 + kc * 32, 256, lane);
      v16bf bb = fragB_pk(w2T, 64, kc, nt * 16, lane);
      acc = wmma_bf16(a, bb, acc);
    }
    int n = nt * 16 + (lane & 15);
    float bv = state_b2[n], gsv = state_gs2[g * 64 + n], gbv = state_gb2[g * 64 + n];
    int mbase = mt * 16 + ((lane >> 4) << 3);
    float colsum = 0.0f;
#pragma unroll
    for (int r = 0; r < 8; ++r) {
      int m = mbase + r;
      float cand = tanhf((acc[r] + bv) * gsv + gbv);
      float dec  = sigm(decay_logit[cell * 64 + m]);
      float hn   = dec * Hp[m * 64 + n] + (1.0f - dec) * cand;
      out_h[cell * 4096 + m * 64 + n] = hn;
      sHnew[m * 64 + n] = (bf16_t)hn;
      colsum += hn;
    }
    atomicAdd(&sHm[n], colsum);
  }
  __syncthreads();

  // Phase 6: msg layer1 (K=64, N=256) from h_new -> sHid
  for (int t = wave; t < 64; t += 8) {
    int mt = t >> 4, nt = t & 15;
    v8f acc = {};
#pragma unroll
    for (int kc = 0; kc < 2; ++kc) {
      v16bf a  = fragA_rm(sHnew + mt * 16 * 64 + kc * 32, 64, lane);
      v16bf bb = fragB_pk(mw1T, 256, kc, nt * 16, lane);
      acc = wmma_bf16(a, bb, acc);
    }
    int n = nt * 16 + (lane & 15);
    float bv = msg_b1[n], gsv = msg_gs1[g * 256 + n], gbv = msg_gb1[g * 256 + n];
    int mbase = mt * 16 + ((lane >> 4) << 3);
#pragma unroll
    for (int r = 0; r < 8; ++r)
      sHid[(mbase + r) * 256 + n] = (bf16_t)tanhf((acc[r] + bv) * gsv + gbv);
  }
  __syncthreads();

  // Phase 7: msg layer2 (K=256, N=64) -> tanh affine + neuron_id -> msg_new, readout
  for (int t = wave; t < 16; t += 8) {
    int mt = t >> 2, nt = t & 3;
    v8f acc = {};
#pragma unroll
    for (int kc = 0; kc < 8; ++kc) {
      v16bf a  = fragA_rm(sHid + mt * 16 * 256 + kc * 32, 256, lane);
      v16bf bb = fragB_pk(mw2T, 64, kc, nt * 16, lane);
      acc = wmma_bf16(a, bb, acc);
    }
    int n = nt * 16 + (lane & 15);
    float bv = msg_b2[n], gsv = msg_gs2[g * 64 + n], gbv = msg_gb2[g * 64 + n];
    int mbase = mt * 16 + ((lane >> 4) << 3);
    float colsum = 0.0f, ro = 0.0f;
#pragma unroll
    for (int r = 0; r < 8; ++r) {
      int m = mbase + r;
      float mv = tanhf((acc[r] + bv) * gsv + gbv) + Np[m * 64 + n];
      out_msg[cell * 4096 + m * 64 + n] = mv;
      colsum += mv;
      if (mt == 0 && lane < 16 && r >= 4 && r < 8) ro += mv;   // rows 4..7
    }
    atomicAdd(&sMm[n], colsum);
    if (mt == 0 && lane < 16)
      out_readout[(size_t)b * (NCC * DNN) + c * 64 + n] = ro * 0.5f;  // ALPHA^-0.5
  }
  __syncthreads();

  // Phase 8: h_mean / msg_mean -> feats workspace (slots 0..127)
  if (tid < 64) {
    float* f = feats_ws + cell * MODIN;
    f[tid]      = sHm[tid] * (1.0f / 64.0f);
    f[64 + tid] = sMm[tid] * (1.0f / 64.0f);
  }
}

// ---------------- per-cell modulation MLP: one block per cell, batch=8 in one M-tile ----
__global__ __launch_bounds__(256) void memgraph_mod_kernel(
    const float* __restrict__ W, const float* __restrict__ decay_logit,
    const float* __restrict__ cell_context, const float* __restrict__ border_gate_logit,
    const float* __restrict__ mod_w1, const float* __restrict__ mod_b1,
    const float* __restrict__ mod_w2, const float* __restrict__ mod_b2,
    const float* __restrict__ feats_ws,
    float* __restrict__ out_cc, float* __restrict__ out_bg) {
  __shared__ bf16_t sFeats[16 * 224];   // K padded 198 -> 224, rows 8..15 zero
  __shared__ bf16_t sHidB[16 * 128];
  __shared__ float  sWr[8];
  __shared__ float  sDm[8];

  const int tid  = threadIdx.x;
  const int lane = tid & 31;
  const int wave = tid >> 5;
  const int c = blockIdx.x;

  for (int i = tid; i < 16 * 224; i += 256) sFeats[i] = (bf16_t)0.0f;
  if (tid < 8) { sWr[tid] = 0.0f; sDm[tid] = 0.0f; }
  __syncthreads();

  // per-batch |W| mean and decay mean (wave w handles batch b=w)
  {
    const size_t cell = (size_t)wave * NCC + c;
    float s = 0.0f;
    for (int i = lane; i < 4096; i += 32) s += fabsf(W[cell * 4096 + i]);
    atomicAdd(&sWr[wave], s);
    float s2 = 0.0f;
    for (int i = lane; i < 64; i += 32) s2 += decay_logit[cell * 64 + i];
    atomicAdd(&sDm[wave], s2);
  }
  __syncthreads();

  // assemble feats rows (batch 0..7)
  for (int i = tid; i < 8 * MODIN; i += 256) {
    int bb = i / MODIN, k = i % MODIN;
    const size_t cell = (size_t)bb * NCC + c;
    float v;
    if (k < 128)       v = feats_ws[cell * MODIN + k];
    else if (k == 128) v = sWr[bb] * (1.0f / 4096.0f);
    else if (k == 129) v = sDm[bb] * (1.0f / 64.0f);
    else if (k < 194)  v = cell_context[cell * 64 + (k - 130)];
    else               v = border_gate_logit[cell * 4 + (k - 194)];
    sFeats[bb * 224 + k] = (bf16_t)v;
  }
  __syncthreads();

  // layer1: [16 x 224] @ [198 x 128] (zero-padded K), wave w -> ntile w
  {
    int nt = wave;
    const float* w1p = mod_w1 + (size_t)c * MODIN * HMODD;
    v8f acc = {};
#pragma unroll
    for (int kc = 0; kc < 7; ++kc) {
      v16bf a  = fragA_rm(sFeats + kc * 32, 224, lane);
      v16bf bb = fragB_f32(w1p, HMODD, kc * 32, nt * 16, MODIN, HMODD, lane);
      acc = wmma_bf16(a, bb, acc);
    }
    int n = nt * 16 + (lane & 15);
    float bv = mod_b1[c * HMODD + n];
    int mbase = (lane >> 4) << 3;
#pragma unroll
    for (int r = 0; r < 8; ++r)
      sHidB[(mbase + r) * 128 + n] = (bf16_t)tanhf(acc[r] + bv);
  }
  __syncthreads();

  // layer2: [16 x 128] @ [128 x 68] (N padded to 80), waves 0..4
  if (wave < 5) {
    int nt = wave;
    const float* w2p = mod_w2 + (size_t)c * HMODD * MODOUT;
    v8f acc = {};
#pragma unroll
    for (int kc = 0; kc < 4; ++kc) {
      v16bf a  = fragA_rm(sHidB + kc * 32, 128, lane);
      v16bf bb = fragB_f32(w2p, MODOUT, kc * 32, nt * 16, HMODD, MODOUT, lane);
      acc = wmma_bf16(a, bb, acc);
    }
    int n = nt * 16 + (lane & 15);
    if (n < MODOUT && (lane >> 4) == 0) {   // valid batch rows m=r (0..7)
#pragma unroll
      for (int r = 0; r < 8; ++r) {
        const size_t cell = (size_t)r * NCC + c;
        float val = acc[r] + mod_b2[c * MODOUT + n];
        if (n < 64) out_cc[cell * 64 + n] = cell_context[cell * 64 + n] + val;
        else        out_bg[cell * 4 + (n - 64)] = border_gate_logit[cell * 4 + (n - 64)] + val;
      }
    }
  }
}

extern "C" void kernel_launch(void* const* d_in, const int* in_sizes, int n_in,
                              void* d_out, int out_size, void* d_ws, size_t ws_size,
                              hipStream_t stream) {
  (void)in_sizes; (void)n_in; (void)out_size; (void)ws_size;
  const float* x    = (const float*)d_in[0];
  const float* h    = (const float*)d_in[1];
  const float* msg  = (const float*)d_in[2];
  const float* W    = (const float*)d_in[3];
  const float* dlg  = (const float*)d_in[4];
  const float* cctx = (const float*)d_in[5];
  const float* bgl  = (const float*)d_in[6];
  const float* nid  = (const float*)d_in[7];
  const float* sw1  = (const float*)d_in[8];
  const float* sb1  = (const float*)d_in[9];
  const float* sgs1 = (const float*)d_in[10];
  const float* sgb1 = (const float*)d_in[11];
  const float* sw2  = (const float*)d_in[12];
  const float* sb2  = (const float*)d_in[13];
  const float* sgs2 = (const float*)d_in[14];
  const float* sgb2 = (const float*)d_in[15];
  const float* mw1  = (const float*)d_in[16];
  const float* mb1  = (const float*)d_in[17];
  const float* mgs1 = (const float*)d_in[18];
  const float* mgb1 = (const float*)d_in[19];
  const float* mw2  = (const float*)d_in[20];
  const float* mb2  = (const float*)d_in[21];
  const float* mgs2 = (const float*)d_in[22];
  const float* mgb2 = (const float*)d_in[23];
  const float* injw = (const float*)d_in[24];
  const float* injb = (const float*)d_in[25];
  const float* modw1 = (const float*)d_in[26];
  const float* modb1 = (const float*)d_in[27];
  const float* modw2 = (const float*)d_in[28];
  const float* modb2 = (const float*)d_in[29];
  const int*   c2g  = (const int*)d_in[30];

  float* out = (float*)d_out;
  float* out_readout = out;                                  // 524288
  float* out_h       = out + 524288;                         // 33554432
  float* out_msg     = out + 524288 + 33554432;              // 33554432
  float* out_cc      = out + 524288 + 2 * 33554432;          // 524288
  float* out_bg      = out + 2 * 524288 + 2 * 33554432;      // 32768

  char* ws = (char*)d_ws;
  float* feats = (float*)ws;
  size_t off = (size_t)BSZ * NCC * MODIN * sizeof(float);    // ~6.49 MB
  bf16_t* w1T  = (bf16_t*)(ws + off); off += 128 * 256 * sizeof(bf16_t);
  bf16_t* w2T  = (bf16_t*)(ws + off); off += 256 * 64  * sizeof(bf16_t);
  bf16_t* mw1T = (bf16_t*)(ws + off); off += 64  * 256 * sizeof(bf16_t);
  bf16_t* mw2T = (bf16_t*)(ws + off); off += 256 * 64  * sizeof(bf16_t);

  memgraph_prep_kernel<<<320, 256, 0, stream>>>(sw1, sw2, mw1, mw2, w1T, w2T, mw1T, mw2T);

  memgraph_cell_kernel<<<BSZ * NCC, 256, 0, stream>>>(
      x, h, msg, W, dlg, bgl, nid,
      sb1, sgs1, sgb1, sb2, sgs2, sgb2,
      mb1, mgs1, mgb1, mb2, mgs2, mgb2,
      injw, injb, c2g, w1T, w2T, mw1T, mw2T,
      out_readout, out_h, out_msg, feats);

  memgraph_mod_kernel<<<NCC, 256, 0, stream>>>(
      W, dlg, cctx, bgl, modw1, modb1, modw2, modb2, feats, out_cc, out_bg);
}
